// Dconv_cos_71408126263594
// MI455X (gfx1250) — matrix operand order repaired and checked by reading, not compiled, not run on hardware
//
#include <hip/hip_runtime.h>
#include <hip/hip_bf16.h>
#include <stdint.h>

typedef __attribute__((ext_vector_type(16))) _Float16 v16h;
typedef __attribute__((ext_vector_type(8)))  _Float16 v8h;
typedef __attribute__((ext_vector_type(8)))  float    v8f;

#define BATCH 8
#define CH    64
#define HH    56
#define WW    56
#define HW    (HH*WW)          // 3136
#define KTOT  576              // 9 * 64, K-dim order kk = k*64 + c
#define NPIX  32               // pixels per block (2 N-tiles)
#define NBLK  98               // 3136 / 32

union V16 { v16h v; v8h h[2]; };

// ---------------------------------------------------------------------------
// Kernel 1: pack weight [64,64,3,3] f32 -> Aprep[o][k*64+c] f16 (row-major K)
// ---------------------------------------------------------------------------
__global__ void prep_weight(const float* __restrict__ w, _Float16* __restrict__ a) {
    int idx = blockIdx.x * 256 + threadIdx.x;
    if (idx >= 64 * KTOT) return;
    int o = idx / KTOT, r = idx % KTOT;
    int k = r >> 6, c = r & 63;                 // kk = k*64 + c
    a[idx] = (_Float16)w[o * KTOT + c * 9 + k]; // weight[o][c][k]
}

// ---------------------------------------------------------------------------
// Kernel 2: one wave32 per (b,p). Lane l<25 owns window slot l.
// Cosine sims via per-lane dot + shfl broadcast of the center vector,
// then 9 rounds of butterfly min-with-index; emit selected slots in
// ascending slot order (== ascending flat index order).
// ---------------------------------------------------------------------------
__global__ void select_k(const float* __restrict__ x, int* __restrict__ sel) {
    int wid  = blockIdx.x * (blockDim.x >> 5) + (threadIdx.x >> 5);
    int lane = threadIdx.x & 31;
    if (wid >= BATCH * HW) return;
    int b = wid / HW, p = wid % HW;
    int i = p / WW, j = p % WW;

    int di = lane / 5 - 2, dj = lane % 5 - 2;
    int wi = i + di, wj = j + dj;
    bool valid = (lane < 25) && wi >= 0 && wi < HH && wj >= 0 && wj < WW;
    int q = valid ? (wi * WW + wj) : p;         // safe address for dead lanes

    const float* xb = x + (size_t)b * CH * HW;
    float dot = 0.f, nq = 0.f;
    for (int c = 0; c < CH; ++c) {
        float xq = xb[c * HW + q];
        float xp = __shfl(xq, 12, 32);          // slot 12 = center pixel
        dot += xp * xq;
        nq  += xq * xq;
    }
    float np  = __shfl(nq, 12, 32);
    float sim = valid ? (dot / sqrtf(np * nq)) : __builtin_inff();

    unsigned mask = 0u;
    for (int r = 0; r < 9; ++r) {
        float s = sim; int id = lane;
        for (int off = 16; off; off >>= 1) {
            float os = __shfl_xor(s, off, 32);
            int   oi = __shfl_xor(id, off, 32);
            if (os < s || (os == s && oi < id)) { s = os; id = oi; }
        }
        mask |= (1u << id);                     // all lanes agree on winner
        if (lane == id) sim = __builtin_inff(); // retire winner
    }

    if (lane == 0) {
        int* dst = sel + ((size_t)b * HW + p) * 9;
        int cnt = 0;
        for (int sl = 0; sl < 25; ++sl)
            if (mask & (1u << sl)) {
                int si = i + sl / 5 - 2, sj = j + sl % 5 - 2;
                dst[cnt++] = si * WW + sj;      // slot order == sorted index order
            }
    }
}

// ---------------------------------------------------------------------------
// Kernel 3: gather-conv as WMMA GEMM.  Block = 128 threads (4 waves) handles
// one (batch, 32-pixel) tile.  Wave w computes output channels [16w,16w+16)
// for BOTH 16-pixel sub-tiles: two accumulators share each A fragment, so
// the weight stream is read once per 32 outputs and the two independent
// accumulation chains keep the XDL pipe fed.  K = 576 in 18 steps.
// sel indices are staged via the CDNA5 async global->LDS path (ASYNCcnt).
// ---------------------------------------------------------------------------
__global__ void __launch_bounds__(128)
conv_wmma(const float* __restrict__ x, const _Float16* __restrict__ A,
          const int* __restrict__ sel, float* __restrict__ out) {
    __shared__ __align__(16) _Float16 G[NPIX * KTOT]; // gathered x, f16, [n][kk]
    __shared__ int S[NPIX * 9];                       // sel for this tile

    int tid = threadIdx.x;
    int b   = blockIdx.y;
    int p0  = blockIdx.x * NPIX;

    // ---- async copy of sel tile into LDS (GLOBAL_LOAD_ASYNC_TO_LDS_B32) ----
    if (tid < NPIX * 9) {
        const int* gsrc = sel + ((size_t)b * HW + p0 + tid / 9) * 9 + tid % 9;
        unsigned ldsoff = (unsigned)(uintptr_t)&S[tid];   // addr[31:0] = LDS offset
        asm volatile("global_load_async_to_lds_b32 %0, %1, off"
                     :: "v"(ldsoff), "v"(gsrc) : "memory");
    }
    asm volatile("s_wait_asynccnt 0x0" ::: "memory");
    __syncthreads();

    const float* xb = x + (size_t)b * CH * HW;
    for (int idx = tid; idx < NPIX * KTOT; idx += 128) {
        int n = idx / KTOT, kk = idx % KTOT;
        int k = kk >> 6, c = kk & 63;
        G[idx] = (_Float16)xb[c * HW + S[n * 9 + k]];
    }
    __syncthreads();

    int wv = tid >> 5, lane = tid & 31;
    int h  = lane >> 4, mn = lane & 15;             // mn = M row (A) and N col (B)

    const _Float16* arow  = A + (size_t)(wv * 16 + mn) * KTOT;
    const _Float16* brow0 = G + mn * KTOT;          // pixels p0 .. p0+15
    const _Float16* brow1 = G + (16 + mn) * KTOT;   // pixels p0+16 .. p0+31

    v8f acc0 = {}, acc1 = {};
    for (int kb = 0; kb < KTOT; kb += 32) {
        V16 af, bf0, bf1;
        // A 16x32 f16, contiguous K order used here: lane 16h+m -> K = kb+16h .. kb+16h+15
        af.h[0] = *(const v8h*)(arow + kb + h * 16);
        af.h[1] = *(const v8h*)(arow + kb + h * 16 + 8);
        // B 32x16 f16, same contiguous K order as A: lane 16h+n -> K = kb+16h .. kb+16h+15
        bf0.h[0] = *(const v8h*)(brow0 + kb + h * 16);
        bf0.h[1] = *(const v8h*)(brow0 + kb + h * 16 + 8);
        bf1.h[0] = *(const v8h*)(brow1 + kb + h * 16);
        bf1.h[1] = *(const v8h*)(brow1 + kb + h * 16 + 8);
        acc0 = __builtin_amdgcn_wmma_f32_16x16x32_f16(
            false, af.v, false, bf0.v, (short)0, acc0, false, false);
        acc1 = __builtin_amdgcn_wmma_f32_16x16x32_f16(
            false, af.v, false, bf1.v, (short)0, acc1, false, false);
    }

    // D layout: VGPR r -> M = r + 8h, N = mn
    float* ob = out + ((size_t)b * CH + wv * 16) * HW + p0 + mn;
    #pragma unroll
    for (int r = 0; r < 8; ++r) {
        ob[(size_t)(r + 8 * h) * HW]      = acc0[r];
        ob[(size_t)(r + 8 * h) * HW + 16] = acc1[r];
    }
}

// ---------------------------------------------------------------------------
extern "C" void kernel_launch(void* const* d_in, const int* in_sizes, int n_in,
                              void* d_out, int out_size, void* d_ws, size_t ws_size,
                              hipStream_t stream) {
    const float* x = (const float*)d_in[0];   // [8,64,56,56]
    const float* w = (const float*)d_in[1];   // [64,64,3,3]
    float* out = (float*)d_out;               // [8,64,56,56]

    _Float16* Aprep = (_Float16*)d_ws;                                     // 73,728 B
    int* sel = (int*)((char*)d_ws + (size_t)64 * KTOT * sizeof(_Float16)); // 903,168 B

    prep_weight<<<(64 * KTOT + 255) / 256, 256, 0, stream>>>(w, Aprep);
    select_k<<<(BATCH * HW) / 8, 256, 0, stream>>>(x, sel);                // 8 waves/block
    conv_wmma<<<dim3(NBLK, BATCH), 128, 0, stream>>>(x, Aprep, sel, out);
}
